// DINLayer_26508538150924
// MI455X (gfx1250) — matrix operand order repaired and checked by reading, not compiled, not run on hardware
//
#include <hip/hip_runtime.h>
#include <math.h>

// ---------------------------------------------------------------------------
// DIN model for MI455X (gfx1250).
//  * structured 2448-wide activation GEMM folded to per-batch M_b[48x36]+c_b
//  * dominant per-batch GEMM: bf16 operands pre-packed in WMMA-native padded
//    layouts -> fragments loaded straight from global as b128, zero LDS;
//    output stored transposed+padded so the epilogue is 2 unguarded b128 stores
//  * MLP GEMMs: LDS-staged bf16 tiles, N-major B so fragments are ds_load_b128
//  * Dice stats / LayerNorm / masked seq-sum / softmax via LDS reductions
// ---------------------------------------------------------------------------

#define Bsz  512
#define Ssz  100
#define SP   112    // Ssz padded to 16-row tiles
#define Dsz  16
#define Fsz  48     // 3*D   (GEMM K)
#define FP   64     // K padded to 32-chunks
#define Hsz  36     // activation-unit hidden (GEMM N)
#define HP   48     // N padded to 16-col tiles
#define CMB  176    // 8*D + F
#define N1   200
#define N2   80

typedef __bf16 bf16x16 __attribute__((ext_vector_type(16)));
typedef float  f32x8   __attribute__((ext_vector_type(8)));

union Frag { bf16x16 v; uint4 q[2]; };
union AccU { f32x8 v; float4 f4[2]; };

// ---------------- weight folding: WA = W[0:48]+W[48:96], WS = W[96:144]-W[48:96]
__global__ void prep_w_kernel(const float* __restrict__ W1,
                              float* __restrict__ WA, float* __restrict__ WS) {
  int idx = blockIdx.x * blockDim.x + threadIdx.x;
  if (idx < Fsz * Hsz) {
    int r = idx / Hsz, k = idx % Hsz;
    float wb = W1[(48 + r) * Hsz + k];
    WA[idx] = W1[r * Hsz + k] + wb;
    WS[idx] = W1[(96 + r) * Hsz + k] - wb;
  }
}

// ---------------- gather profile (8 embeds -> xcomb[:,0:128]) and v_item
__global__ void gather_profile_kernel(const int* __restrict__ uid, const int* __restrict__ t1,
                                      const int* __restrict__ t2, const int* __restrict__ t3,
                                      const int* __restrict__ t4, const int* __restrict__ gid,
                                      const int* __restrict__ sid, const int* __restrict__ cid,
                                      const float* __restrict__ emb,
                                      float* __restrict__ xcomb, float* __restrict__ vitem) {
  int b = blockIdx.x, t = threadIdx.x;
  if (t >= 176) return;
  int e, d, dst_item = -1, dst_prof = -1;
  if (t < 128) { e = t >> 4; d = t & 15; dst_prof = t; }
  else         { int u = t - 128; e = 5 + (u >> 4); d = u & 15; dst_item = u; }
  int idx;
  switch (e) {
    case 0: idx = uid[b]; break;
    case 1: idx = t1[b];  break;
    case 2: idx = t2[b];  break;
    case 3: idx = t3[b];  break;
    case 4: idx = t4[b];  break;
    case 5: idx = gid[b]; break;
    case 6: idx = sid[b]; break;
    default: idx = cid[b]; break;
  }
  float v = emb[idx * Dsz + d];
  if (dst_prof >= 0) xcomb[b * CMB + dst_prof] = v;
  else               vitem[b * Fsz + dst_item] = v;
}

// ---------------- gather behavior series -> bf16, WMMA-A padded layout [B][SP][FP]
__global__ void gather_vser_kernel(const int* __restrict__ vg, const int* __restrict__ vs,
                                   const int* __restrict__ vc, const float* __restrict__ emb,
                                   __bf16* __restrict__ vserp) {
  int t = blockIdx.x * blockDim.x + threadIdx.x;
  if (t >= Bsz * Ssz * FP) return;
  int j = t % FP, bs = t / FP;           // bs = b*Ssz + s
  int b = bs / Ssz, s = bs % Ssz;
  float v = 0.f;
  if (j < Fsz) {                          // K-padding (j>=48) must be zero
    int f = j >> 4, d = j & 15;
    const int* tab = (f == 0) ? vg : ((f == 1) ? vs : vc);
    v = emb[tab[bs] * Dsz + d];
  }
  vserp[((size_t)b * SP + s) * FP + j] = (__bf16)v;
}

// ---------------- per-batch fold:  MbT[b][h][i] = WS[i,h] + sum_j vitem_j*Wouter[i,j,h]
//                  (N-major, K-padded bf16)    cvecp = [vitem @ WA + b_act1, 0-pad to 48]
__global__ void mb_kernel(const float* __restrict__ vitem, const float* __restrict__ W1,
                          const float* __restrict__ WA, const float* __restrict__ WS,
                          const float* __restrict__ b1,
                          __bf16* __restrict__ MbT, float* __restrict__ cvecp) {
  int b = blockIdx.x, t = threadIdx.x;
  __shared__ float vit[Fsz];
  if (t < Fsz) vit[t] = vitem[b * Fsz + t];
  __syncthreads();
  if (t < HP) {
    float c = 0.f;
    if (t < Hsz) {
      c = b1[t];
      for (int j = 0; j < Fsz; ++j) c += vit[j] * WA[j * Hsz + t];
    }
    cvecp[b * HP + t] = c;
  }
  for (int o = t; o < Hsz * FP; o += blockDim.x) {
    int h = o >> 6, i = o & 63;          // h = hidden (N), i = contraction (K)
    float acc = 0.f;
    if (i < Fsz) {
      acc = WS[i * Hsz + h];
      const float* wrow = W1 + (144 + i * Fsz) * Hsz + h;   // outer block at row 144
      #pragma unroll 4
      for (int j = 0; j < Fsz; ++j) acc += vit[j] * wrow[j * Hsz];
    }
    MbT[((size_t)b * HP + h) * FP + i] = (__bf16)acc;
  }
}

// ---------------- dominant GEMM: hpre[b] = v_series[b](100x48) @ M_b(48x36) + c_b
// One wave per 16x16 tile; fragments loaded as b128 straight from the packed
// bf16 global buffers (no LDS).  Output stored TRANSPOSED + PADDED:
// hpreT[b][n(48)][m(112)], so each lane's 8 accumulator values (fixed column,
// 8 consecutive rows) become two unguarded global_store_b128.
__global__ void hpre_gemm_kernel(const __bf16* __restrict__ vserp,
                                 const __bf16* __restrict__ MbT,
                                 const float* __restrict__ cvecp,
                                 float* __restrict__ hpreT) {
  const int b = blockIdx.z, tm = blockIdx.x, tn = blockIdx.y;
  const int lane = threadIdx.x & 31;
  const int m0 = lane & 15, half = lane >> 4;
  const __bf16* arow = vserp + ((size_t)b * SP + tm * 16 + m0) * FP;
  const __bf16* brow = MbT   + ((size_t)b * HP + tn * 16 + m0) * FP;
  f32x8 acc = {0.f, 0.f, 0.f, 0.f, 0.f, 0.f, 0.f, 0.f};
  #pragma unroll
  for (int k0 = 0; k0 < FP; k0 += 32) {
    Frag fa, fb;
    // A 16x32 bf16 fragment: lanes 0-15 hold K {0..7,16..23}, lanes 16-31 {8..15,24..31}
    fa.q[0] = *(const uint4*)(arow + k0 + half * 8);
    fa.q[1] = *(const uint4*)(arow + k0 + 16 + half * 8);
    // B 32x16 bf16 fragment (N-major rows): lanes 0-15 K {0..7,16..23}, lanes 16-31 {8..15,24..31}
    fb.q[0] = *(const uint4*)(brow + k0 + half * 8);
    fb.q[1] = *(const uint4*)(brow + k0 + 16 + half * 8);
    acc = __builtin_amdgcn_wmma_f32_16x16x32_bf16(false, fa.v, false, fb.v,
                                                  (short)0, acc, false, false);
  }
  const int gn  = tn * 16 + m0;            // output column (< 48, padded)
  const int gm0 = tm * 16 + 8 * half;      // first of 8 consecutive output rows
  const float bias_v = cvecp[b * HP + gn]; // zero in padded columns
  AccU u;
  #pragma unroll
  for (int r = 0; r < 8; ++r) u.v[r] = acc[r] + bias_v;
  float* crow = hpreT + ((size_t)b * HP + gn) * SP + gm0;
  *(float4*)(crow)     = u.f4[0];
  *(float4*)(crow + 4) = u.f4[1];
}

// ---------------- generic single-wave WMMA GEMM (f32 in, bf16 tiles via LDS)
// M must be a multiple of 16 (M = 512 here); only the N edge is guarded.
__global__ void gemm_bias_kernel(const float* __restrict__ A, int lda,
                                 const float* __restrict__ W, int ldw,
                                 const float* __restrict__ bias,
                                 float* __restrict__ C, int ldc,
                                 int M, int N, int K) {
  __shared__ alignas(16) __bf16 As[16][32];
  __shared__ alignas(16) __bf16 BsT[16][32];      // N-major: BsT[n][k]
  const int tm = blockIdx.x, tn = blockIdx.y;
  const int lane = threadIdx.x & 31;
  const int m0 = lane & 15, half = lane >> 4;
  f32x8 acc = {0.f, 0.f, 0.f, 0.f, 0.f, 0.f, 0.f, 0.f};

  for (int k0 = 0; k0 < K; k0 += 32) {
    for (int i = threadIdx.x; i < 16 * 32; i += blockDim.x) {
      int r = i >> 5, c = i & 31;
      int gk = k0 + c;
      float v = (gk < K) ? A[(size_t)(tm * 16 + r) * lda + gk] : 0.f;
      As[r][c] = (__bf16)v;
    }
    for (int i = threadIdx.x; i < 32 * 16; i += blockDim.x) {
      int r = i >> 4, c = i & 15;                 // r = k, c = n
      int gk = k0 + r, gn = tn * 16 + c;
      float v = (gk < K && gn < N) ? W[(size_t)gk * ldw + gn] : 0.f;
      BsT[c][r] = (__bf16)v;
    }
    __syncthreads();
    Frag fa, fb;
    fa.q[0] = *(const uint4*)&As[m0][half * 8];
    fa.q[1] = *(const uint4*)&As[m0][16 + half * 8];
    fb.q[0] = *(const uint4*)&BsT[m0][half * 8];
    fb.q[1] = *(const uint4*)&BsT[m0][16 + half * 8];
    acc = __builtin_amdgcn_wmma_f32_16x16x32_bf16(false, fa.v, false, fb.v,
                                                  (short)0, acc, false, false);
    __syncthreads();
  }
  const int gn = tn * 16 + m0;
  if (gn < N) {
    const float bias_v = bias[gn];
    #pragma unroll
    for (int r = 0; r < 8; ++r) {
      int gm = tm * 16 + r + 8 * half;
      C[(size_t)gm * ldc + gn] = acc[r] + bias_v;
    }
  }
}

// ---------------- batch stats for activation Dice over hpreT[b][k][s]
__global__ void stats_act_kernel(const float* __restrict__ hpreT,
                                 float* __restrict__ mu, float* __restrict__ rinv) {
  int col = blockIdx.x, t = threadIdx.x;          // col = s*36 + k
  int s = col / Hsz, k = col % Hsz;
  const float* p = hpreT + (size_t)k * SP + s;
  __shared__ float ssum[128], ssq[128];
  float sum = 0.f, q = 0.f;
  for (int r = t; r < Bsz; r += blockDim.x) {
    float v = p[(size_t)r * (HP * SP)];
    sum += v; q += v * v;
  }
  ssum[t] = sum; ssq[t] = q; __syncthreads();
  for (int off = 64; off > 0; off >>= 1) {
    if (t < off) { ssum[t] += ssum[t + off]; ssq[t] += ssq[t + off]; }
    __syncthreads();
  }
  if (t == 0) {
    float m = ssum[0] / Bsz;
    float v = ssq[0] / Bsz - m * m;
    mu[col] = m;
    rinv[col] = rsqrtf(v + 1e-3f);
  }
}

// ---------------- per-channel batch stats (mean, rsqrt(var+eps)) over rows
__global__ void stats_kernel(const float* __restrict__ src, int rows, int ld,
                             float* __restrict__ mu, float* __restrict__ rinv) {
  int col = blockIdx.x, t = threadIdx.x;
  __shared__ float ssum[128], ssq[128];
  float s = 0.f, q = 0.f;
  for (int r = t; r < rows; r += blockDim.x) {
    float v = src[(size_t)r * ld + col];
    s += v; q += v * v;
  }
  ssum[t] = s; ssq[t] = q; __syncthreads();
  for (int off = 64; off > 0; off >>= 1) {
    if (t < off) { ssum[t] += ssum[t + off]; ssq[t] += ssq[t + off]; }
    __syncthreads();
  }
  if (t == 0) {
    float m = ssum[0] / rows;
    float v = ssq[0] / rows - m * m;
    mu[col] = m;
    rinv[col] = rsqrtf(v + 1e-3f);
  }
}

// ---------------- Dice(hpre) -> scores -> mask -> x_inter (xcomb[:,128:176])
__global__ void act_combine_kernel(const float* __restrict__ hpreT, const float* __restrict__ muA,
                                   const float* __restrict__ rinvA, const float* __restrict__ alpha,
                                   const float* __restrict__ Wout, const float* __restrict__ bout,
                                   const int* __restrict__ visg, const __bf16* __restrict__ vserp,
                                   float* __restrict__ xcomb) {
  int b = blockIdx.x, t = threadIdx.x;
  __shared__ float sc[Ssz];
  if (t < Ssz) {
    int s = t;
    const float* hp = hpreT + (size_t)b * HP * SP + s;   // stride SP over k
    const float* m  = muA + s * Hsz;
    const float* rv = rinvA + s * Hsz;
    float acc = 0.f;
    #pragma unroll 4
    for (int k = 0; k < Hsz; ++k) {
      float x  = hp[k * SP];
      float xn = (x - m[k]) * rv[k];
      float p  = 1.f / (1.f + expf(-xn));
      float h  = alpha[k] * (1.f - p) * x + p * x;
      acc += h * Wout[k];
    }
    acc += bout[0];
    float mask = (visg[b * Ssz + s] == 0) ? 1.f : 0.f;   // reference keeps PADDING positions
    sc[s] = acc * mask;
  }
  __syncthreads();
  if (t < Fsz) {
    const __bf16* vp = vserp + (size_t)b * SP * FP + t;
    float x = 0.f;
    for (int s = 0; s < Ssz; ++s) x += sc[s] * (float)vp[s * FP];
    xcomb[b * CMB + 128 + t] = x;
  }
}

// ---------------- per-row LayerNorm (in place)
__global__ void ln_kernel(float* __restrict__ x, const float* __restrict__ g,
                          const float* __restrict__ beta, int C) {
  int row = blockIdx.x, t = threadIdx.x;
  __shared__ float ssum[256], ssq[256];
  float* xr = x + (size_t)row * C;
  float s = 0.f, q = 0.f;
  for (int c = t; c < C; c += blockDim.x) { float v = xr[c]; s += v; q += v * v; }
  ssum[t] = s; ssq[t] = q; __syncthreads();
  for (int off = 128; off > 0; off >>= 1) {
    if (t < off) { ssum[t] += ssum[t + off]; ssq[t] += ssq[t + off]; }
    __syncthreads();
  }
  float m  = ssum[0] / C;
  float vv = ssq[0] / C - m * m;
  float rs = rsqrtf(vv + 1e-3f);
  for (int c = t; c < C; c += blockDim.x) xr[c] = (xr[c] - m) * rs * g[c] + beta[c];
}

// ---------------- Dice apply (in place) with precomputed batch stats
__global__ void dice_kernel(float* __restrict__ x, const float* __restrict__ mu,
                            const float* __restrict__ rinv, const float* __restrict__ alpha,
                            int C, int n) {
  int i = blockIdx.x * blockDim.x + threadIdx.x;
  if (i >= n) return;
  int c = i % C;
  float v  = x[i];
  float xn = (v - mu[c]) * rinv[c];
  float p  = 1.f / (1.f + expf(-xn));
  x[i] = alpha[c] * (1.f - p) * v + p * v;
}

// ---------------- logits + softmax
__global__ void out_kernel(const float* __restrict__ d2, const float* __restrict__ Wout,
                           const float* __restrict__ bout, float* __restrict__ out) {
  int b = blockIdx.x, t = threadIdx.x;
  __shared__ float s0[128], s1[128];
  float p0 = 0.f, p1 = 0.f;
  if (t < N2) {
    float v = d2[b * N2 + t];
    p0 = v * Wout[t * 2];
    p1 = v * Wout[t * 2 + 1];
  }
  s0[t] = p0; s1[t] = p1; __syncthreads();
  for (int off = 64; off > 0; off >>= 1) {
    if (t < off) { s0[t] += s0[t + off]; s1[t] += s1[t + off]; }
    __syncthreads();
  }
  if (t == 0) {
    float l0 = s0[0] + bout[0], l1 = s1[0] + bout[1];
    float m  = fmaxf(l0, l1);
    float e0 = expf(l0 - m), e1 = expf(l1 - m);
    float inv = 1.f / (e0 + e1);
    out[b * 2]     = e0 * inv;
    out[b * 2 + 1] = e1 * inv;
  }
}

// ---------------------------------------------------------------------------
extern "C" void kernel_launch(void* const* d_in, const int* in_sizes, int n_in,
                              void* d_out, int out_size, void* d_ws, size_t ws_size,
                              hipStream_t stream) {
  const int* uid = (const int*)d_in[0];
  const int* t1  = (const int*)d_in[1];
  const int* t2  = (const int*)d_in[2];
  const int* t3  = (const int*)d_in[3];
  const int* t4  = (const int*)d_in[4];
  const int* gid = (const int*)d_in[5];
  const int* sid = (const int*)d_in[6];
  const int* cid = (const int*)d_in[7];
  const int* vg  = (const int*)d_in[8];
  const int* vs  = (const int*)d_in[9];
  const int* vc  = (const int*)d_in[10];
  const float* emb      = (const float*)d_in[11];
  const float* W_act1   = (const float*)d_in[12];
  const float* b_act1   = (const float*)d_in[13];
  const float* a_act    = (const float*)d_in[14];
  const float* W_actout = (const float*)d_in[15];
  const float* b_actout = (const float*)d_in[16];
  const float* W_mlp1   = (const float*)d_in[17];
  const float* b_mlp1   = (const float*)d_in[18];
  const float* g_ln1    = (const float*)d_in[19];
  const float* be_ln1   = (const float*)d_in[20];
  const float* a_mlp1   = (const float*)d_in[21];
  const float* W_mlp2   = (const float*)d_in[22];
  const float* b_mlp2   = (const float*)d_in[23];
  const float* g_ln2    = (const float*)d_in[24];
  const float* be_ln2   = (const float*)d_in[25];
  const float* a_mlp2   = (const float*)d_in[26];
  const float* W_out    = (const float*)d_in[27];
  const float* b_out    = (const float*)d_in[28];
  float* out = (float*)d_out;

  // ---- workspace layout: f32 region first, then 16B-aligned bf16 region (~23 MB)
  float* w = (float*)d_ws;
  size_t o = 0;
  float* WA    = w + o; o += Fsz * Hsz;
  float* WS    = w + o; o += Fsz * Hsz;
  float* vitem = w + o; o += (size_t)Bsz * Fsz;
  float* xcomb = w + o; o += (size_t)Bsz * CMB;
  float* cvecp = w + o; o += (size_t)Bsz * HP;
  float* hpreT = w + o; o += (size_t)Bsz * HP * SP;   // transposed + padded
  float* muA   = w + o; o += Ssz * Hsz;
  float* rinvA = w + o; o += Ssz * Hsz;
  float* h1    = w + o; o += (size_t)Bsz * N1;        // h1 -> ln1 -> dice1 in place
  float* mu1   = w + o; o += N1;
  float* rinv1 = w + o; o += N1;
  float* h2    = w + o; o += (size_t)Bsz * N2;
  float* mu2   = w + o; o += N2;
  float* rinv2 = w + o; o += N2;
  o = (o + 7) & ~(size_t)7;                           // 16B-align the bf16 region
  __bf16* vserp = (__bf16*)(w + o); o += ((size_t)Bsz * SP * FP) / 2;
  __bf16* MbT   = (__bf16*)(w + o); o += ((size_t)Bsz * HP * FP) / 2;

  // 1. fold activation-unit weights
  prep_w_kernel<<<7, 256, 0, stream>>>(W_act1, WA, WS);
  // 2. gathers (embedding table lives in L2: 10.2 MB of 192 MB)
  gather_profile_kernel<<<Bsz, 192, 0, stream>>>(uid, t1, t2, t3, t4, gid, sid, cid,
                                                 emb, xcomb, vitem);
  gather_vser_kernel<<<(Bsz * Ssz * FP + 255) / 256, 256, 0, stream>>>(vg, vs, vc, emb, vserp);
  // 3. per-batch fold M_b (bf16, N-major, K-padded), c_b (padded to 48)
  mb_kernel<<<Bsz, 192, 0, stream>>>(vitem, W_act1, WA, WS, b_act1, MbT, cvecp);
  // 4. hpre = v_series @ M_b + c_b  (WMMA, b128 fragments from global, b128 stores)
  hpre_gemm_kernel<<<dim3(SP / 16, HP / 16, Bsz), 32, 0, stream>>>(vserp, MbT, cvecp, hpreT);
  // 5. Dice batch stats over B for each (s,k) channel
  stats_act_kernel<<<Ssz * Hsz, 128, 0, stream>>>(hpreT, muA, rinvA);
  // 6. dice -> scores -> padding mask -> weighted sequence sum into xcomb[:,128:176]
  act_combine_kernel<<<Bsz, 128, 0, stream>>>(hpreT, muA, rinvA, a_act, W_actout, b_actout,
                                              vg, vserp, xcomb);
  // 7. MLP1 (WMMA) + LN + Dice
  gemm_bias_kernel<<<dim3(Bsz / 16, (N1 + 15) / 16), 32, 0, stream>>>(
      xcomb, CMB, W_mlp1, N1, b_mlp1, h1, N1, Bsz, N1, CMB);
  ln_kernel<<<Bsz, 256, 0, stream>>>(h1, g_ln1, be_ln1, N1);
  stats_kernel<<<N1, 128, 0, stream>>>(h1, Bsz, N1, mu1, rinv1);
  dice_kernel<<<(Bsz * N1 + 255) / 256, 256, 0, stream>>>(h1, mu1, rinv1, a_mlp1, N1, Bsz * N1);
  // 8. MLP2 (WMMA) + LN + Dice
  gemm_bias_kernel<<<dim3(Bsz / 16, N2 / 16), 32, 0, stream>>>(
      h1, N1, W_mlp2, N2, b_mlp2, h2, N2, Bsz, N2, N1);
  ln_kernel<<<Bsz, 256, 0, stream>>>(h2, g_ln2, be_ln2, N2);
  stats_kernel<<<N2, 128, 0, stream>>>(h2, Bsz, N2, mu2, rinv2);
  dice_kernel<<<(Bsz * N2 + 255) / 256, 256, 0, stream>>>(h2, mu2, rinv2, a_mlp2, N2, Bsz * N2);
  // 9. logits + softmax
  out_kernel<<<Bsz, 128, 0, stream>>>(h2, W_out, b_out, out);
}
